// SwinTransformerBlock_correct_85761906967112
// MI455X (gfx1250) — matrix operand order repaired and checked by reading, not compiled, not run on hardware
//
#include <hip/hip_runtime.h>
#include <hip/hip_bf16.h>
#include <math.h>

// ---------------------------------------------------------------------------
// Types / helpers
// ---------------------------------------------------------------------------
typedef __attribute__((ext_vector_type(16))) __bf16          v16bf;
typedef __attribute__((ext_vector_type(8)))  float           v8f;
typedef __attribute__((ext_vector_type(8)))  unsigned short  ushort8;
typedef __attribute__((ext_vector_type(16))) unsigned short  ushort16;
typedef __attribute__((ext_vector_type(4)))  float           float4v;

typedef int i32x4v __attribute__((vector_size(16)));
typedef __attribute__((address_space(1))) i32x4v as1_i32x4;
typedef __attribute__((address_space(3))) i32x4v as3_i32x4;

#define DEV __device__ __forceinline__

#if defined(__gfx1250__) && __has_builtin(__builtin_amdgcn_global_load_async_to_lds_b128)
#define HAS_ASYNC_LDS 1
#else
#define HAS_ASYNC_LDS 0
#endif

DEV unsigned short f2bf(float f) {
  union { float f; unsigned u; } v; v.f = f;
  unsigned u = v.u;
  u += 0x7fffu + ((u >> 16) & 1u);          // round-to-nearest-even
  return (unsigned short)(u >> 16);
}
DEV float bf2f(unsigned short h) {
  union { unsigned u; float f; } v; v.u = ((unsigned)h) << 16;
  return v.f;
}

// 16-byte global -> LDS copy; async DMA path on CDNA5 when available.
DEV void cp16_g2l(const unsigned short* gsrc, unsigned short* ldst) {
#if HAS_ASYNC_LDS
  __builtin_amdgcn_global_load_async_to_lds_b128(
      (as1_i32x4*)gsrc, (as3_i32x4*)ldst, 0, 0);
#else
  *(ushort8*)ldst = *(const ushort8*)gsrc;
#endif
}
DEV void cp16_wait() {
#if HAS_ASYNC_LDS && __has_builtin(__builtin_amdgcn_s_wait_asynccnt)
  __builtin_amdgcn_s_wait_asynccnt(0);
#endif
}

union FragCast { ushort16 u; v16bf b; };

// A-operand fragment (16x32 bf16): lane holds row (lane&15);
// elems 0..7 -> k = kb + hi*8 + e ; elems 8..15 -> k = kb + 16 + hi*8 + (e-8)
DEV v16bf ldfragA(const unsigned short* p, int row, int stride, int kb, int hi) {
  const unsigned short* q = p + row * stride + kb + hi * 8;
  ushort8 a = *(const ushort8*)q;
  ushort8 b = *(const ushort8*)(q + 16);
  FragCast f;
  f.u = __builtin_shufflevector(a, b, 0,1,2,3,4,5,6,7,8,9,10,11,12,13,14,15);
  return f.b;
}
// B-operand fragment (32x16 bf16, stored as [n][k] in LDS): lane holds col (lane&15);
// elems 0..7 -> k = kb + hi*8 + e ; elems 8..15 -> k = kb + 16 + hi*8 + (e-8)
DEV v16bf ldfragB(const unsigned short* p, int row, int stride, int kb, int hi) {
  const unsigned short* q = p + row * stride + kb + hi * 8;
  ushort8 a = *(const ushort8*)q;
  ushort8 b = *(const ushort8*)(q + 16);
  FragCast f;
  f.u = __builtin_shufflevector(a, b, 0,1,2,3,4,5,6,7,8,9,10,11,12,13,14,15);
  return f.b;
}

DEV v8f wmma_bf16(v16bf a, v16bf b, v8f c) {
  return __builtin_amdgcn_wmma_f32_16x16x32_bf16(false, a, false, b, (short)0, c,
                                                 false, false);
}

DEV float gelu_exact(float x) { return 0.5f * x * (1.f + erff(x * 0.70710678118654752f)); }

// ---------------------------------------------------------------------------
// Weight conversion kernels
// ---------------------------------------------------------------------------
__global__ void cvt_plain_k(const float* __restrict__ s, unsigned short* __restrict__ d, int n) {
  int i = blockIdx.x * 256 + threadIdx.x;
  if (i < n) d[i] = f2bf(s[i]);
}

// (Cout,Cin,KH,KW) f32 -> [ky][kx][cout][cin] bf16
__global__ void cvt_convw_k(const float* __restrict__ s, unsigned short* __restrict__ d,
                            int Cout, int Cin, int KH, int KW) {
  int i = blockIdx.x * 256 + threadIdx.x;
  int total = Cout * Cin * KH * KW;
  if (i >= total) return;
  int ci = i % Cin; int t = i / Cin;
  int co = t % Cout; t /= Cout;
  int kx = t % KW; int ky = t / KW;
  d[i] = f2bf(s[((co * Cin + ci) * KH + ky) * KW + kx]);
}

// ---------------------------------------------------------------------------
// Layout / elementwise kernels
// ---------------------------------------------------------------------------
__global__ void nchw2nhwc_k(const float* __restrict__ s, float* __restrict__ dst,
                            int outStride, int outOff) {
  long long i = (long long)blockIdx.x * 256 + threadIdx.x;
  if (i >= 200704LL * 64) return;
  int c = (int)(i & 63); long long pix = i >> 6;
  long long b = pix / 50176; long long rem = pix % 50176;
  dst[pix * (long long)outStride + outOff + c] = s[(b * 64 + c) * 50176 + rem];
}

__global__ void mul_k(const float* __restrict__ a, float* __restrict__ b, long long n) {
  long long i = (long long)blockIdx.x * 256 + threadIdx.x;
  if (i < n) b[i] = a[i] * b[i];
}

__global__ void fma3_k(const float* __restrict__ a, const float* __restrict__ b,
                       const float* __restrict__ c, float* __restrict__ d, long long n) {
  long long i = (long long)blockIdx.x * 256 + threadIdx.x;
  if (i < n) d[i] = a[i] * b[i] + c[i];
}

// window_partition (rev=0) / window_reverse (rev=1), C=64, WS=14, H=W=224
__global__ void winperm_k(const float* __restrict__ src, float* __restrict__ dst, int rev) {
  long long i = (long long)blockIdx.x * 256 + threadIdx.x;
  if (i >= 200704LL * 64) return;
  int c = (int)(i & 63); long long pix = i >> 6;
  int wi = (int)(pix / 196), n = (int)(pix % 196);
  int b = wi >> 8, hw = (wi >> 4) & 15, ww = wi & 15;
  int r = n / 14, s2 = n % 14;
  long long npix = ((long long)(b * 224 + hw * 14 + r)) * 224 + ww * 14 + s2;
  if (rev) dst[npix * 64 + c] = src[pix * 64 + c];
  else     dst[pix * 64 + c]  = src[npix * 64 + c];
}

// layernorm over C=64; one 32-lane wave per row
__global__ __launch_bounds__(256) void ln64_k(const float* __restrict__ src, int stride,
                                              const float* __restrict__ g,
                                              const float* __restrict__ bta,
                                              float* __restrict__ dst, int M) {
  int row = blockIdx.x * 8 + (threadIdx.x >> 5);
  int lane = threadIdx.x & 31;
  if (row >= M) return;
  const float* p = src + (size_t)row * stride;
  float x0 = p[lane], x1 = p[lane + 32];
  float s = x0 + x1;
#pragma unroll
  for (int o = 16; o > 0; o >>= 1) s += __shfl_xor(s, o, 32);
  float mean = s * (1.f / 64.f);
  float d0 = x0 - mean, d1 = x1 - mean;
  float v = d0 * d0 + d1 * d1;
#pragma unroll
  for (int o = 16; o > 0; o >>= 1) v += __shfl_xor(v, o, 32);
  float rs = rsqrtf(v * (1.f / 64.f) + 1e-5f);
  dst[(size_t)row * 64 + lane]      = d0 * rs * g[lane] + bta[lane];
  dst[(size_t)row * 64 + lane + 32] = d1 * rs * g[lane + 32] + bta[lane + 32];
}

// out = NCHW(sx + h2)
__global__ void final_k(const float* __restrict__ sx, const float* __restrict__ h2,
                        float* __restrict__ out) {
  long long i = (long long)blockIdx.x * 256 + threadIdx.x;
  if (i >= 200704LL * 64) return;
  long long b = i / (64LL * 50176);
  long long c = (i / 50176) % 64;
  long long rem = i % 50176;
  long long pix = b * 50176 + rem;
  out[i] = sx[pix * 64 + c] + h2[pix * 64 + c];
}

// ---------------------------------------------------------------------------
// Implicit-GEMM conv, NHWC, bf16 WMMA.  One block = one output row (224 px).
// NT = Cout/16 (4 or 8).  KS = kernel size (square, pad=(KS-1)/2).
// Haloed A-tile loaded once per (ky, kc); all KS weight tiles staged via
// async global->LDS; kx tap loop runs entirely out of LDS (fragment row shift).
// act: 0 none, 1 relu, 2 sigmoid(relu(x))
// ---------------------------------------------------------------------------
template <int NT, int KS>
__global__ __launch_bounds__(256) void conv_wmma_k(
    const float* __restrict__ in, int cinStride, int cinOff,
    const unsigned short* __restrict__ wt, const float* __restrict__ bias,
    float* __restrict__ out, int coutStride, int coutOff,
    int Cin, int act) {
  constexpr int PW   = (14 * NT) / 8;        // accum tiles per wave
  constexpr int PAD  = (KS - 1) / 2;
  constexpr int HALO = 224 + KS - 1;         // haloed row length
  constexpr int Cout = NT * 16;
  __shared__ unsigned short As[HALO * 32];
  __shared__ unsigned short Bs[KS * Cout * 32];
  const int bh = blockIdx.x;
  const int b = bh / 224, h = bh % 224;
  const int tid = threadIdx.x, wave = tid >> 5, lane = tid & 31;
  const int ln = lane & 15, hi = lane >> 4;

  v8f acc[PW];
#pragma unroll
  for (int t = 0; t < PW; ++t) acc[t] = v8f{};

  for (int ky = 0; ky < KS; ++ky) {
    int iy = h + ky - PAD;
    if (iy < 0 || iy >= 224) continue;       // uniform across block
    for (int kc = 0; kc < Cin; kc += 32) {
      // ---- haloed A tile: HALO x 32, f32 -> bf16 ----
      for (int g = tid; g < (HALO * 32) / 4; g += 256) {
        int hx = g >> 3, c = (g & 7) << 2;
        int ix = hx - PAD;
        float4v v = float4v{};
        if (ix >= 0 && ix < 224) {
          const float* sp = in + (size_t)((b * 224 + iy) * 224 + ix) * cinStride +
                            cinOff + kc + c;
          v = *(const float4v*)sp;
          if (kc + 32 < Cin) __builtin_prefetch(sp + 32, 0, 0);
        }
        unsigned short* dp = As + hx * 32 + c;
        dp[0] = f2bf(v[0]); dp[1] = f2bf(v[1]); dp[2] = f2bf(v[2]); dp[3] = f2bf(v[3]);
      }
      // ---- all KS weight tiles (bf16, async DMA to LDS) ----
      for (int g = tid; g < KS * NT * 64; g += 256) {
        int row = g >> 2, c = (g & 3) << 3;  // row = kx*Cout + n
        cp16_g2l(wt + ((size_t)ky * KS * Cout + row) * Cin + kc + c,
                 Bs + row * 32 + c);
      }
      cp16_wait();
      __syncthreads();
#pragma unroll
      for (int kx = 0; kx < KS; ++kx) {
#pragma unroll
        for (int t = 0; t < PW; ++t) {
          int tt = wave * PW + t;
          int i = tt / NT, j = tt % NT;
          v16bf af  = ldfragA(As, i * 16 + ln + kx, 32, 0, hi);
          v16bf bf_ = ldfragB(Bs + kx * Cout * 32, j * 16 + ln, 32, 0, hi);
          acc[t] = wmma_bf16(af, bf_, acc[t]);
        }
      }
      __syncthreads();
    }
  }
#pragma unroll
  for (int t = 0; t < PW; ++t) {
    int tt = wave * PW + t;
    int i = tt / NT, j = tt % NT;
    int n = j * 16 + ln;
    float bv = bias[n];
#pragma unroll
    for (int r = 0; r < 8; ++r) {
      int m = i * 16 + hi * 8 + r;
      float val = acc[t][r] + bv;
      if (act == 1) val = fmaxf(val, 0.f);
      else if (act == 2) { val = fmaxf(val, 0.f); val = 1.f / (1.f + expf(-val)); }
      out[(size_t)((b * 224 + h) * 224 + m) * coutStride + coutOff + n] = val;
    }
  }
}

// ---------------------------------------------------------------------------
// GEMM: D[M,N] = act(A[M,K] @ W^T + bias),  W bf16 [N][K].  act: 0 none, 1 gelu
// Block tile 128x64, 8 waves in 4x2, each wave 2x2 WMMA tiles.
// ---------------------------------------------------------------------------
__global__ __launch_bounds__(256) void gemm_k(
    const float* __restrict__ A, const unsigned short* __restrict__ Bw,
    const float* __restrict__ bias, float* __restrict__ D,
    int M, int N, int K, int act) {
  __shared__ unsigned short As[128 * 32];
  __shared__ unsigned short Bs[64 * 32];
  const int bm = blockIdx.x * 128, bn = blockIdx.y * 64;
  const int tid = threadIdx.x, wave = tid >> 5, lane = tid & 31;
  const int ln = lane & 15, hi = lane >> 4;
  const int wm = (wave >> 1) * 32, wn = (wave & 1) * 32;

  v8f acc[2][2];
#pragma unroll
  for (int a = 0; a < 2; ++a)
#pragma unroll
    for (int b = 0; b < 2; ++b) acc[a][b] = v8f{};

  for (int kc = 0; kc < K; kc += 32) {
    for (int g = tid; g < 1024; g += 256) {
      int m = g >> 3, c = (g & 7) << 2;
      float4v v = float4v{};
      if (bm + m < M) {
        const float* sp = A + (size_t)(bm + m) * K + kc + c;
        v = *(const float4v*)sp;
        if (kc + 32 < K) __builtin_prefetch(sp + 32, 0, 0);
      }
      unsigned short* dp = As + m * 32 + c;
      dp[0] = f2bf(v[0]); dp[1] = f2bf(v[1]); dp[2] = f2bf(v[2]); dp[3] = f2bf(v[3]);
    }
    {
      int g = tid;                           // 256 groups exactly
      int n = g >> 2, c = (g & 3) << 3;
      cp16_g2l(Bw + (size_t)(bn + n) * K + kc + c, Bs + n * 32 + c);
    }
    cp16_wait();
    __syncthreads();
#pragma unroll
    for (int si = 0; si < 2; ++si) {
      v16bf af = ldfragA(As, wm + si * 16 + ln, 32, 0, hi);
#pragma unroll
      for (int sj = 0; sj < 2; ++sj) {
        v16bf bf_ = ldfragB(Bs, wn + sj * 16 + ln, 32, 0, hi);
        acc[si][sj] = wmma_bf16(af, bf_, acc[si][sj]);
      }
    }
    __syncthreads();
  }
#pragma unroll
  for (int si = 0; si < 2; ++si)
#pragma unroll
    for (int sj = 0; sj < 2; ++sj) {
      int n = bn + wn + sj * 16 + ln;
      float bv = bias[n];
#pragma unroll
      for (int r = 0; r < 8; ++r) {
        int m = bm + wm + si * 16 + hi * 8 + r;
        if (m < M) {
          float val = acc[si][sj][r] + bv;
          if (act == 1) val = gelu_exact(val);
          D[(size_t)m * N + n] = val;
        }
      }
    }
}

// ---------------------------------------------------------------------------
// Fused window attention.  One block per (window, head).
// scores = [q*scale | xbw] @ [k | corrW]^T  (single K=32 WMMA per tile)
// softmax (f32 math, bf16 storage), out = attn @ v via WMMA (K padded to 224).
// Dynamic LDS: attn 208x224 bf16 + A' 208x32 + B' 208x32 + vT 16x224.
// ---------------------------------------------------------------------------
__global__ __launch_bounds__(256) void attn_k(
    const float* __restrict__ qkv,     // (1024*196, 192)  [q|k|v], bias applied
    const float* __restrict__ xbw,     // (1024*196, 64)
    const unsigned short* __restrict__ corrW,  // (196,16) bf16
    const float* __restrict__ corrB,   // (196)
    float* __restrict__ outp)          // (1024*196, 64)
{
  extern __shared__ unsigned short smem_attn[];
  unsigned short* attnS = smem_attn;            // 208*224
  unsigned short* Ap = attnS + 208 * 224;       // 208*32
  unsigned short* Bp = Ap + 208 * 32;           // 208*32
  unsigned short* vT = Bp + 208 * 32;           // 16*224
  const int wi = blockIdx.x, hd = blockIdx.y;
  const int tid = threadIdx.x, wave = tid >> 5, lane = tid & 31;
  const int ln = lane & 15, hi = lane >> 4;
  const float scale = 0.25f;                    // 16^-0.5

  for (int idx = tid; idx < 208 * 32; idx += 256) {
    int m = idx >> 5, k = idx & 31;
    float a = 0.f; unsigned short bb = 0;
    if (m < 196) {
      size_t row = (size_t)(wi * 196 + m);
      if (k < 16) {
        a  = qkv[row * 192 + hd * 16 + k] * scale;
        bb = f2bf(qkv[row * 192 + 64 + hd * 16 + k]);
      } else {
        a  = xbw[row * 64 + hd * 16 + (k - 16)];
        bb = corrW[m * 16 + (k - 16)];
      }
    }
    Ap[idx] = f2bf(a);
    Bp[idx] = bb;
  }
  for (int idx = tid; idx < 16 * 224; idx += 256) {
    int d = idx / 224, n = idx % 224;
    float v = 0.f;
    if (n < 196) v = qkv[(size_t)(wi * 196 + n) * 192 + 128 + hd * 16 + d];
    vT[idx] = f2bf(v);
  }
  __syncthreads();

  // ---- scores: 13x13 tiles over 8 waves ----
  for (int t = wave; t < 169; t += 8) {
    int i = t / 13, j = t % 13;
    v16bf af  = ldfragA(Ap, i * 16 + ln, 32, 0, hi);
    v16bf bf_ = ldfragB(Bp, j * 16 + ln, 32, 0, hi);
    v8f c = v8f{};
    c = wmma_bf16(af, bf_, c);
    int n = j * 16 + ln;
    float cb = (n < 196) ? corrB[n] : 0.f;
#pragma unroll
    for (int r = 0; r < 8; ++r) {
      int m = i * 16 + hi * 8 + r;
      float v = (n < 196) ? (c[r] + cb) : -1e30f;
      attnS[m * 224 + n] = f2bf(v);
    }
  }
  __syncthreads();

  // ---- softmax over 196 real columns; zero padding ----
  if (tid < 196) {
    unsigned short* p = attnS + tid * 224;
    float mx = -1e30f;
    for (int n = 0; n < 196; ++n) mx = fmaxf(mx, bf2f(p[n]));
    float s = 0.f;
    for (int n = 0; n < 196; ++n) { float e = expf(bf2f(p[n]) - mx); s += e; p[n] = f2bf(e); }
    float rinv = 1.f / s;
    for (int n = 0; n < 196; ++n) p[n] = f2bf(bf2f(p[n]) * rinv);
    for (int n = 196; n < 224; ++n) p[n] = 0;
  } else if (tid < 208) {
    unsigned short* p = attnS + tid * 224;
    for (int n = 0; n < 224; ++n) p[n] = 0;
  }
  __syncthreads();

  // ---- out = attn @ v : 13 M-tiles, K = 224 (padded) ----
  for (int i = wave; i < 13; i += 8) {
    v8f c = v8f{};
#pragma unroll
    for (int kc = 0; kc < 224; kc += 32) {
      v16bf af  = ldfragA(attnS, i * 16 + ln, 224, kc, hi);
      v16bf bf_ = ldfragB(vT, ln, 224, kc, hi);
      c = wmma_bf16(af, bf_, c);
    }
#pragma unroll
    for (int r = 0; r < 8; ++r) {
      int m = i * 16 + hi * 8 + r;
      if (m < 196)
        outp[(size_t)(wi * 196 + m) * 64 + hd * 16 + ln] = c[r];
    }
  }
}

// ---------------------------------------------------------------------------
// Host orchestration
// ---------------------------------------------------------------------------
extern "C" void kernel_launch(void* const* d_in, const int* in_sizes, int n_in,
                              void* d_out, int out_size, void* d_ws, size_t ws_size,
                              hipStream_t stream) {
  (void)in_sizes; (void)n_in; (void)out_size;
  const float* x      = (const float*)d_in[0];
  const float* xbasic = (const float*)d_in[1];
  const float* c1w = (const float*)d_in[2];  const float* c1b = (const float*)d_in[3];
  const float* c2w = (const float*)d_in[4];  const float* c2b = (const float*)d_in[5];
  const float* c3w = (const float*)d_in[6];  const float* c3b = (const float*)d_in[7];
  const float* c4w = (const float*)d_in[8];  const float* c4b = (const float*)d_in[9];
  const float* c5w = (const float*)d_in[10]; const float* c5b = (const float*)d_in[11];
  const float* c6w = (const float*)d_in[12]; const float* c6b = (const float*)d_in[13];
  const float* c7w = (const float*)d_in[14]; const float* c7b = (const float*)d_in[15];
  const float* c8w = (const float*)d_in[16]; const float* c8b = (const float*)d_in[17];
  const float* q1w = (const float*)d_in[18]; const float* q1b = (const float*)d_in[19];
  const float* n1g = (const float*)d_in[20]; const float* n1b = (const float*)d_in[21];
  const float* qkvw = (const float*)d_in[22]; const float* qkvb = (const float*)d_in[23];
  const float* corw = (const float*)d_in[24]; const float* corb = (const float*)d_in[25];
  const float* prjw = (const float*)d_in[26]; const float* prjb = (const float*)d_in[27];
  const float* n2g = (const float*)d_in[28]; const float* n2b = (const float*)d_in[29];
  const float* f1w = (const float*)d_in[30]; const float* f1b = (const float*)d_in[31];
  const float* f2w = (const float*)d_in[32]; const float* f2b = (const float*)d_in[33];

  float* ws = (float*)d_ws;
  unsigned short* wsu = (unsigned short*)d_ws;
  const size_t U = 12845056;       // 200704 * 64 floats
  const size_t BASE = 1048576;     // 4MB weight arena (floats)
  if (ws_size < (BASE + 13 * U) * sizeof(float)) return;

  float* regA = ws + BASE;           // 2u : cat12 -> f1
  float* regB = ws + BASE + 2 * U;   // 3u : cat123 -> qkvb -> f2
  float* regC = ws + BASE + 5 * U;   // 1u : xo -> n2
  float* regD = ws + BASE + 6 * U;   // 2u : b1 -> xw, xbw
  float* regE = ws + BASE + 8 * U;   // 2u : b2/b1b2 -> attnout, projw
  float* regF = ws + BASE + 10 * U;  // 1u : b1c -> xtn
  float* regG = ws + BASE + 11 * U;  // 1u : c8o -> xc
  float* regH = ws + BASE + 12 * U;  // 1u : xb -> sx

  // bf16 weight slots (ushort offsets inside arena)
  unsigned short* wc1 = wsu + 0;        // 36864
  unsigned short* wc2 = wsu + 36864;    // 102400
  unsigned short* wc3 = wsu + 139264;   // 200704
  unsigned short* wc4 = wsu + 339968;   // 110592
  unsigned short* wc5 = wsu + 450560;   // 147456
  unsigned short* wc6 = wsu + 598016;   // 147456
  unsigned short* wc7 = wsu + 745472;   // 8192
  unsigned short* wc8 = wsu + 753664;   // 8192
  unsigned short* wq1 = wsu + 761856;   // 36864
  unsigned short* wqk = wsu + 798720;   // 12288
  unsigned short* wco = wsu + 811008;   // 3136
  unsigned short* wpj = wsu + 814144;   // 4096
  unsigned short* wf1 = wsu + 818240;   // 8192
  unsigned short* wf2 = wsu + 826432;   // 8192

  dim3 blk(256);
  auto g1 = [](long long n) { return dim3((unsigned)((n + 255) / 256)); };
  const long long P64 = 200704LL * 64, P128 = 200704LL * 128;

  // ---- weight conversion ----
  cvt_convw_k<<<g1(36864),  blk, 0, stream>>>(c1w, wc1, 64, 64, 3, 3);
  cvt_convw_k<<<g1(102400), blk, 0, stream>>>(c2w, wc2, 64, 64, 5, 5);
  cvt_convw_k<<<g1(200704), blk, 0, stream>>>(c3w, wc3, 64, 64, 7, 7);
  cvt_convw_k<<<g1(110592), blk, 0, stream>>>(c4w, wc4, 64, 192, 3, 3);
  cvt_convw_k<<<g1(147456), blk, 0, stream>>>(c5w, wc5, 128, 128, 3, 3);
  cvt_convw_k<<<g1(147456), blk, 0, stream>>>(c6w, wc6, 128, 128, 3, 3);
  cvt_convw_k<<<g1(8192),   blk, 0, stream>>>(c7w, wc7, 64, 128, 1, 1);
  cvt_convw_k<<<g1(8192),   blk, 0, stream>>>(c8w, wc8, 64, 128, 1, 1);
  cvt_convw_k<<<g1(36864),  blk, 0, stream>>>(q1w, wq1, 64, 64, 3, 3);
  cvt_plain_k<<<g1(12288),  blk, 0, stream>>>(qkvw, wqk, 12288);
  cvt_plain_k<<<g1(3136),   blk, 0, stream>>>(corw, wco, 3136);
  cvt_plain_k<<<g1(4096),   blk, 0, stream>>>(prjw, wpj, 4096);
  cvt_plain_k<<<g1(8192),   blk, 0, stream>>>(f1w, wf1, 8192);
  cvt_plain_k<<<g1(8192),   blk, 0, stream>>>(f2w, wf2, 8192);

  // ---- NCHW -> NHWC cat12 = [x | x_basic] ----
  nchw2nhwc_k<<<g1(P64), blk, 0, stream>>>(x,      regA, 128, 0);
  nchw2nhwc_k<<<g1(P64), blk, 0, stream>>>(xbasic, regA, 128, 64);

  // ---- Aware_offset ----
  conv_wmma_k<4,3><<<896, blk, 0, stream>>>(regA, 128, 0, wc1, c1b, regB, 192, 0,   64, 1);
  conv_wmma_k<4,5><<<896, blk, 0, stream>>>(regA, 128, 0, wc2, c2b, regB, 192, 64,  64, 1);
  conv_wmma_k<4,7><<<896, blk, 0, stream>>>(regA, 128, 0, wc3, c3b, regB, 192, 128, 64, 1);
  conv_wmma_k<4,3><<<896, blk, 0, stream>>>(regB, 192, 0, wc4, c4b, regC, 64, 0,   192, 1);
  conv_wmma_k<8,3><<<896, blk, 0, stream>>>(regA, 128, 0, wc5, c5b, regD, 128, 0,  128, 1);
  conv_wmma_k<8,3><<<896, blk, 0, stream>>>(regD, 128, 0, wc6, c6b, regE, 128, 0,  128, 2);
  mul_k<<<g1(P128), blk, 0, stream>>>(regD, regE, P128);                        // b1*b2
  conv_wmma_k<4,1><<<896, blk, 0, stream>>>(regE, 128, 0, wc7, c7b, regF, 64, 0, 128, 1);
  conv_wmma_k<4,1><<<896, blk, 0, stream>>>(regA, 128, 0, wc8, c8b, regG, 64, 0, 128, 1);
  fma3_k<<<g1(P64), blk, 0, stream>>>(regC, regF, regG, regH, P64);             // xb

  // ---- norm1 + qkv1 conv ----
  ln64_k<<<dim3(200704 / 8), blk, 0, stream>>>(regA, 128, n1g, n1b, regF, 200704);
  conv_wmma_k<4,3><<<896, blk, 0, stream>>>(regF, 64, 0, wq1, q1b, regG, 64, 0, 64, 0);

  // ---- windows + qkv linear ----
  winperm_k<<<g1(P64), blk, 0, stream>>>(regG, regD, 0);          // xw
  winperm_k<<<g1(P64), blk, 0, stream>>>(regH, regD + U, 0);      // xbw
  gemm_k<<<dim3(1568, 3), blk, 0, stream>>>(regD, wqk, qkvb, regB, 200704, 192, 64, 0);

  // ---- fused window attention ----
  attn_k<<<dim3(1024, 4), blk, 126976, stream>>>(regB, regD + U, wco, corb, regE);

  // ---- proj + reverse + MLP ----
  gemm_k<<<dim3(1568, 1), blk, 0, stream>>>(regE, wpj, prjb, regE + U, 200704, 64, 64, 0);
  winperm_k<<<g1(P64), blk, 0, stream>>>(regE + U, regH, 1);      // sx (= xt)
  ln64_k<<<dim3(200704 / 8), blk, 0, stream>>>(regH, 64, n2g, n2b, regC, 200704);
  gemm_k<<<dim3(1568, 2), blk, 0, stream>>>(regC, wf1, f1b, regA, 200704, 128, 64, 1);
  gemm_k<<<dim3(1568, 1), blk, 0, stream>>>(regA, wf2, f2b, regB, 200704, 64, 128, 0);
  final_k<<<g1(P64), blk, 0, stream>>>(regH, regB, (float*)d_out);
}
